// DIT_49881750176273
// MI455X (gfx1250) — matrix operand order repaired and checked by reading, not compiled, not run on hardware
//
#include <hip/hip_runtime.h>
#include <hip/hip_bf16.h>
#include <math.h>

typedef _Float16 v16h __attribute__((ext_vector_type(16)));
typedef _Float16 v8h  __attribute__((ext_vector_type(8)));
typedef float    v8f  __attribute__((ext_vector_type(8)));

union F16x16 { v16h v; v8h h8[2]; _Float16 h[16]; };

#define S_TOK 2048
#define N_TOK 1024
#define DMODEL 1024
#define NHEAD 16
#define HDIM 64
#define DH4 4096

static __device__ __forceinline__ v8f wmma_f16(const F16x16& a, const F16x16& b, v8f c) {
    return __builtin_amdgcn_wmma_f32_16x16x32_f16(false, a.v, false, b.v, (short)0, c, false, false);
}

// async 16B/lane copy: global (saddr + voffset) -> LDS[ldsaddr], ASYNCcnt-tracked
static __device__ __forceinline__ void async_copy_b128(unsigned lds_addr, unsigned goff,
                                                       unsigned long long gbase) {
    asm volatile("global_load_async_to_lds_b128 %0, %1, %2"
                 :: "v"(lds_addr), "v"(goff), "s"(gbase) : "memory");
}
static __device__ __forceinline__ void async_wait0() {
    asm volatile("s_wait_asynccnt 0x0" ::: "memory");
}

// ---------------------------------------------------------------- adaLN mods
__global__ void adaln_kernel(const float* __restrict__ c, const float* __restrict__ w,
                             const float* __restrict__ b, float* __restrict__ mods) {
    int j = blockIdx.x * blockDim.x + threadIdx.x;
    if (j >= 6 * DMODEL) return;
    float acc = b[j];
    const float* wr = w + (size_t)j * 128;
    #pragma unroll 8
    for (int k = 0; k < 128; ++k) acc += c[k] * wr[k];
    mods[j] = acc;
}

// ---------------------------------------------------------------- f32 -> f16
__global__ void cvt_kernel(const float* __restrict__ in, _Float16* __restrict__ out, int n) {
    for (int i = blockIdx.x * blockDim.x + threadIdx.x; i < n; i += gridDim.x * blockDim.x)
        out[i] = (_Float16)in[i];
}

// ------------------------------------------------- LayerNorm + modulate -> f16
__global__ __launch_bounds__(256) void ln_mod_kernel(const float* __restrict__ x,
                                                     const float* __restrict__ w,
                                                     const float* __restrict__ mods,
                                                     int shIdx, int scIdx,
                                                     _Float16* __restrict__ out) {
    __shared__ float rs[256], rq[256];
    int row = blockIdx.x, tid = threadIdx.x;
    const float* xr = x + (size_t)row * DMODEL;
    float v[4], s = 0.f, sq = 0.f;
    #pragma unroll
    for (int i = 0; i < 4; ++i) { v[i] = xr[tid + i * 256]; s += v[i]; sq += v[i] * v[i]; }
    rs[tid] = s; rq[tid] = sq; __syncthreads();
    for (int st = 128; st > 0; st >>= 1) {
        if (tid < st) { rs[tid] += rs[tid + st]; rq[tid] += rq[tid + st]; }
        __syncthreads();
    }
    float mean = rs[0] * (1.f / DMODEL);
    float var = rq[0] * (1.f / DMODEL) - mean * mean;
    float rstd = rsqrtf(var + 1e-5f);
    const float* sh = mods + (size_t)shIdx * DMODEL;
    const float* sc = mods + (size_t)scIdx * DMODEL;
    _Float16* orow = out + (size_t)row * DMODEL;
    #pragma unroll
    for (int i = 0; i < 4; ++i) {
        int col = tid + i * 256;
        float h = (v[i] - mean) * rstd * w[col] * (1.f + sc[col]) + sh[col];
        orow[col] = (_Float16)h;
    }
}

// ---------------------------------------------------------------- GEMM + epilogue
// C[M x Nout] = A[M x K](f16 rowmajor) * W[Nout x K]^T (f16 rowmajor), f32 accum.
// Block: 256 thr = 8 waves; block tile 64x128; wave tile 16x64; K-step 32.
// Double-buffered LDS staging via global_load_async_to_lds_b128 (ASYNCcnt),
// single barrier per K-step, copy(k+1) overlapped with compute(k).
template <int MODE>
__global__ __launch_bounds__(256) void gemm_wmma(const _Float16* __restrict__ A,
                                                 const _Float16* __restrict__ W,
                                                 int K,
                                                 const float* __restrict__ mods,
                                                 const float* __restrict__ resid,
                                                 const float* __restrict__ bias,
                                                 const float* __restrict__ cosT,
                                                 const float* __restrict__ sinT,
                                                 _Float16* __restrict__ out_q,
                                                 _Float16* __restrict__ out_k,
                                                 _Float16* __restrict__ out_vT,
                                                 float* __restrict__ out_f32,
                                                 _Float16* __restrict__ out_f16) {
    __shared__ _Float16 As[2][64 * 40];
    __shared__ _Float16 Bs[2][128 * 40];
    const int tid = threadIdx.x;
    const int wave = tid >> 5, lane = tid & 31;
    const int half = lane >> 4, ln16 = lane & 15;
    const int wm = wave & 3, wn = wave >> 2;
    const int m0 = blockIdx.x * 64, n0 = blockIdx.y * 128;

    const unsigned long long Abase = (unsigned long long)(uintptr_t)A;
    const unsigned long long Wbase = (unsigned long long)(uintptr_t)W;

    // per-thread staging chunks (byte LDS addrs are loop-invariant)
    const int arow = tid >> 2, ac8 = (tid & 3) * 8;
    const int brow0 = (tid * 2) >> 2, bc80 = ((tid * 2) & 3) * 8;
    const int brow1 = (tid * 2 + 1) >> 2, bc81 = ((tid * 2 + 1) & 3) * 8;
    unsigned lds_a[2], lds_b0[2], lds_b1[2];
    #pragma unroll
    for (int b = 0; b < 2; ++b) {
        lds_a[b]  = (unsigned)(uintptr_t)&As[b][arow * 40 + ac8];
        lds_b0[b] = (unsigned)(uintptr_t)&Bs[b][brow0 * 40 + bc80];
        lds_b1[b] = (unsigned)(uintptr_t)&Bs[b][brow1 * 40 + bc81];
    }
    const unsigned ga_off = (unsigned)(((m0 + arow) * K + ac8) * 2);
    const unsigned gb0_off = (unsigned)(((n0 + brow0) * K + bc80) * 2);
    const unsigned gb1_off = (unsigned)(((n0 + brow1) * K + bc81) * 2);

    auto issue = [&](int buf, int k0) {
        const unsigned kb = (unsigned)(k0 * 2);
        async_copy_b128(lds_a[buf],  ga_off + kb,  Abase);
        async_copy_b128(lds_b0[buf], gb0_off + kb, Wbase);
        async_copy_b128(lds_b1[buf], gb1_off + kb, Wbase);
    };

    v8f acc[4] = {v8f{}, v8f{}, v8f{}, v8f{}};

    issue(0, 0);
    for (int k0 = 0; k0 < K; k0 += 32) {
        const int cur = (k0 >> 5) & 1;
        async_wait0();        // my tile-k copies are in LDS
        __syncthreads();      // everyone's are; everyone done reading the other buffer
        if (k0 + 32 < K) issue(cur ^ 1, k0 + 32);  // overlap next copy with compute

        F16x16 a;
        const _Float16* ar = &As[cur][(wm * 16 + ln16) * 40];
        a.h8[0] = *(const v8h*)(ar + half * 8);
        a.h8[1] = *(const v8h*)(ar + 16 + half * 8);
        #pragma unroll
        for (int j = 0; j < 4; ++j) {
            F16x16 b;
            const _Float16* br = &Bs[cur][(wn * 64 + j * 16 + ln16) * 40];
            b.h8[0] = *(const v8h*)(br + half * 8);
            b.h8[1] = *(const v8h*)(br + 16 + half * 8);
            acc[j] = wmma_f16(a, b, acc[j]);
        }
    }

    const int mbase = m0 + wm * 16 + 8 * half;
    const int nbase = n0 + wn * 64;

    if (MODE == 0) {
        // rotary epilogue; nbase is 64-aligned -> one (which, head) per wave
        const int which = nbase >> 10;
        const int head = (nbase & 1023) >> 6;
        #pragma unroll
        for (int j = 0; j < 2; ++j) {
            const int dlo = j * 16 + ln16;
            #pragma unroll
            for (int r = 0; r < 8; ++r) {
                int s = mbase + r;
                int pos = s & (N_TOK - 1);
                float cs = cosT[pos * 32 + dlo], sn = sinT[pos * 32 + dlo];
                float xl = acc[j][r], xh = acc[j + 2][r];
                float olo = xl * cs - xh * sn;
                float ohi = xh * cs + xl * sn;
                if (which == 0) {
                    _Float16* p = out_q + ((size_t)head * S_TOK + s) * HDIM;
                    p[dlo] = (_Float16)olo; p[dlo + 32] = (_Float16)ohi;
                } else if (which == 1) {
                    _Float16* p = out_k + ((size_t)head * S_TOK + s) * HDIM;
                    p[dlo] = (_Float16)olo; p[dlo + 32] = (_Float16)ohi;
                } else {
                    out_vT[((size_t)head * HDIM + dlo) * S_TOK + s] = (_Float16)olo;
                    out_vT[((size_t)head * HDIM + dlo + 32) * S_TOK + s] = (_Float16)ohi;
                }
            }
        }
    } else {
        #pragma unroll
        for (int j = 0; j < 4; ++j) {
            int n = nbase + j * 16 + ln16;
            #pragma unroll
            for (int r = 0; r < 8; ++r) {
                int m = mbase + r;
                float v = acc[j][r];
                if (MODE == 1) {
                    out_f32[(size_t)m * DMODEL + n] =
                        mods[2 * DMODEL + n] * v + resid[(size_t)m * DMODEL + n];
                } else if (MODE == 2) {
                    float t = v + bias[n];
                    float g = 0.5f * t * (1.f + tanhf(0.7978845608f * (t + 0.044715f * t * t * t)));
                    out_f16[(size_t)m * DH4 + n] = (_Float16)g;
                } else { // MODE 3
                    float t = v + bias[n];
                    out_f32[(size_t)m * DMODEL + n] =
                        mods[5 * DMODEL + n] * t + resid[(size_t)m * DMODEL + n];
                }
            }
        }
    }
}

// ---------------------------------------------------------------- attention
// One wave per (head, 16-query tile). Block-diff mask is tile-uniform (BS=16).
__global__ __launch_bounds__(256) void attn_kernel(const _Float16* __restrict__ q16,
                                                   const _Float16* __restrict__ k16,
                                                   const _Float16* __restrict__ vT16,
                                                   _Float16* __restrict__ out) {
    __shared__ _Float16 pb[8][16 * 40];
    const int tid = threadIdx.x;
    const int wave = tid >> 5, lane = tid & 31;
    const int half = lane >> 4, ln16 = lane & 15;
    const int gw = blockIdx.x * 8 + wave;
    const int head = gw >> 7, qt = gw & 127;

    // Q fragments (A-layout), K=0..31 and 32..63
    const _Float16* qrow = q16 + ((size_t)head * S_TOK + qt * 16 + ln16) * HDIM;
    F16x16 a_lo, a_hi;
    a_lo.h8[0] = *(const v8h*)(qrow + half * 8);
    a_lo.h8[1] = *(const v8h*)(qrow + 16 + half * 8);
    a_hi.h8[0] = *(const v8h*)(qrow + 32 + half * 8);
    a_hi.h8[1] = *(const v8h*)(qrow + 48 + half * 8);

    // all-ones B fragment: rowsum(P) via WMMA (layout-agnostic)
    F16x16 bones;
    #pragma unroll
    for (int i = 0; i < 16; ++i) bones.h[i] = (_Float16)1.0f;

    float mr[8], lr[8];
    v8f o[4] = {v8f{}, v8f{}, v8f{}, v8f{}};
    #pragma unroll
    for (int r = 0; r < 8; ++r) { mr[r] = -3.0e38f; lr[r] = 0.f; }

    const int cnt = (qt < 64) ? (qt + 1) : (qt - 63);
    _Float16* pw = pb[wave];

    for (int i = 0; i < cnt; i += 2) {
        const int kt0 = (qt < 64) ? ((i == 0) ? qt : 63 + i) : (64 + i);
        const bool v1 = (i + 1) < cnt;
        const int i1 = i + 1;
        const int kt1 = v1 ? ((qt < 64) ? ((i1 == 0) ? qt : 63 + i1) : (64 + i1)) : kt0;

        // S tiles: Q(16x64) x K^T(64x16) per key tile
        v8f sA = v8f{}, sB = v8f{};
        {
            const _Float16* kr = k16 + ((size_t)head * S_TOK + kt0 * 16 + ln16) * HDIM;
            F16x16 b;
            b.h8[0] = *(const v8h*)(kr + half * 8);
            b.h8[1] = *(const v8h*)(kr + 16 + half * 8);
            sA = wmma_f16(a_lo, b, sA);
            b.h8[0] = *(const v8h*)(kr + 32 + half * 8);
            b.h8[1] = *(const v8h*)(kr + 48 + half * 8);
            sA = wmma_f16(a_hi, b, sA);
        }
        {
            const _Float16* kr = k16 + ((size_t)head * S_TOK + kt1 * 16 + ln16) * HDIM;
            F16x16 b;
            b.h8[0] = *(const v8h*)(kr + half * 8);
            b.h8[1] = *(const v8h*)(kr + 16 + half * 8);
            sB = wmma_f16(a_lo, b, sB);
            b.h8[0] = *(const v8h*)(kr + 32 + half * 8);
            b.h8[1] = *(const v8h*)(kr + 48 + half * 8);
            sB = wmma_f16(a_hi, b, sB);
        }

        // online softmax: row max via 16-lane butterfly; row sum via ones-WMMA
        float pA[8], pB[8], alpha[8];
        #pragma unroll
        for (int r = 0; r < 8; ++r) {
            float sa = sA[r] * 0.125f;
            float sb = v1 ? sB[r] * 0.125f : -3.0e38f;
            float mx = fmaxf(sa, sb);
            #pragma unroll
            for (int off = 1; off < 16; off <<= 1) mx = fmaxf(mx, __shfl_xor(mx, off, 32));
            float mn = fmaxf(mr[r], mx);
            float al = __expf(mr[r] - mn);
            pA[r] = __expf(sa - mn);
            pB[r] = v1 ? __expf(sb - mn) : 0.f;
            mr[r] = mn;
            alpha[r] = al;
            lr[r] *= al;
        }
        #pragma unroll
        for (int t = 0; t < 4; ++t)
            #pragma unroll
            for (int r = 0; r < 8; ++r) o[t][r] *= alpha[r];

        // P: D-layout -> A-layout via private LDS patch (wave-local, no barrier)
        #pragma unroll
        for (int r = 0; r < 8; ++r) {
            int m = r + 8 * half;
            pw[m * 40 + ln16] = (_Float16)pA[r];
            pw[m * 40 + 16 + ln16] = (_Float16)pB[r];
        }
        F16x16 pf;
        const _Float16* pr = pw + ln16 * 40;
        pf.h8[0] = *(const v8h*)(pr + half * 16);
        pf.h8[1] = *(const v8h*)(pr + half * 16 + 8);

        // row sums of P in one WMMA: D(m,n) = sum_k P[m,k] for every n
        v8f rs8 = wmma_f16(pf, bones, v8f{});
        #pragma unroll
        for (int r = 0; r < 8; ++r) lr[r] += rs8[r];

        // O += P(16x32) x V(32x64); V^T rows give contiguous key runs
        const int kth = half ? kt1 : kt0;
        #pragma unroll
        for (int t = 0; t < 4; ++t) {
            F16x16 vb;
            const _Float16* vr = vT16 + ((size_t)head * HDIM + t * 16 + ln16) * S_TOK + kth * 16;
            vb.h8[0] = *(const v8h*)(vr);
            vb.h8[1] = *(const v8h*)(vr + 8);
            o[t] = wmma_f16(pf, vb, o[t]);
        }
    }

    // write O / l as f16 into [s][head*64 + d]
    #pragma unroll
    for (int t = 0; t < 4; ++t) {
        #pragma unroll
        for (int r = 0; r < 8; ++r) {
            int s = qt * 16 + r + 8 * half;
            int d = t * 16 + ln16;
            float ov = o[t][r] / lr[r];
            out[(size_t)s * DMODEL + head * HDIM + d] = (_Float16)ov;
        }
    }
}

// ---------------------------------------------------------------- launch
extern "C" void kernel_launch(void* const* d_in, const int* in_sizes, int n_in,
                              void* d_out, int out_size, void* d_ws, size_t ws_size,
                              hipStream_t stream) {
    const float* x        = (const float*)d_in[0];
    const float* c        = (const float*)d_in[1];
    const float* cosT     = (const float*)d_in[2];
    const float* sinT     = (const float*)d_in[3];
    const float* norm1_w  = (const float*)d_in[4];
    const float* qkv_w    = (const float*)d_in[5];
    const float* attn_w   = (const float*)d_in[6];
    const float* norm2_w  = (const float*)d_in[7];
    const float* mlp_w1   = (const float*)d_in[8];
    const float* mlp_b1   = (const float*)d_in[9];
    const float* mlp_w2   = (const float*)d_in[10];
    const float* mlp_b2   = (const float*)d_in[11];
    const float* adaLN_w  = (const float*)d_in[12];
    const float* adaLN_b  = (const float*)d_in[13];
    float* out = (float*)d_out;

    char* base = (char*)d_ws;
    size_t off = 0;
    auto take = [&](size_t bytes) -> char* {
        char* p = base + off;
        off += (bytes + 255) & ~(size_t)255;
        return p;
    };
    float*     mods   = (float*)    take((size_t)6 * DMODEL * 4);
    _Float16*  h16    = (_Float16*) take((size_t)S_TOK * DMODEL * 2);
    _Float16*  wq16   = (_Float16*) take((size_t)3 * DMODEL * DMODEL * 2);
    _Float16*  wa16   = (_Float16*) take((size_t)DMODEL * DMODEL * 2);
    _Float16*  w1_16  = (_Float16*) take((size_t)DH4 * DMODEL * 2);
    _Float16*  w2_16  = (_Float16*) take((size_t)DMODEL * DH4 * 2);
    _Float16*  q16    = (_Float16*) take((size_t)NHEAD * S_TOK * HDIM * 2);
    _Float16*  k16    = (_Float16*) take((size_t)NHEAD * S_TOK * HDIM * 2);
    _Float16*  vT16   = (_Float16*) take((size_t)NHEAD * HDIM * S_TOK * 2);
    _Float16*  ai16   = (_Float16*) take((size_t)S_TOK * DMODEL * 2);
    float*     x2     = (float*)    take((size_t)S_TOK * DMODEL * 4);
    _Float16*  h2_16  = (_Float16*) take((size_t)S_TOK * DMODEL * 2);
    _Float16*  m1_16  = (_Float16*) take((size_t)S_TOK * DH4 * 2);
    (void)ws_size; (void)in_sizes; (void)n_in; (void)out_size;

    // adaLN modulation vectors
    adaln_kernel<<<24, 256, 0, stream>>>(c, adaLN_w, adaLN_b, mods);

    // weight conversions f32 -> f16
    cvt_kernel<<<1024, 256, 0, stream>>>(qkv_w,  wq16,  3 * DMODEL * DMODEL);
    cvt_kernel<<<1024, 256, 0, stream>>>(attn_w, wa16,  DMODEL * DMODEL);
    cvt_kernel<<<1024, 256, 0, stream>>>(mlp_w1, w1_16, DH4 * DMODEL);
    cvt_kernel<<<1024, 256, 0, stream>>>(mlp_w2, w2_16, DMODEL * DH4);

    // LN1 + modulate(sh_msa=0, sc_msa=1)
    ln_mod_kernel<<<S_TOK, 256, 0, stream>>>(x, norm1_w, mods, 0, 1, h16);

    // QKV GEMM + rotary
    gemm_wmma<0><<<dim3(32, 24), 256, 0, stream>>>(h16, wq16, DMODEL,
        nullptr, nullptr, nullptr, cosT, sinT, q16, k16, vT16, nullptr, nullptr);

    // masked flash attention
    attn_kernel<<<256, 256, 0, stream>>>(q16, k16, vT16, ai16);

    // output projection: x2 = g_msa * (o @ Wo^T) + x
    gemm_wmma<1><<<dim3(32, 8), 256, 0, stream>>>(ai16, wa16, DMODEL,
        mods, x, nullptr, nullptr, nullptr, nullptr, nullptr, nullptr, x2, nullptr);

    // LN2 + modulate(sh_mlp=3, sc_mlp=4)
    ln_mod_kernel<<<S_TOK, 256, 0, stream>>>(x2, norm2_w, mods, 3, 4, h2_16);

    // MLP1 + gelu
    gemm_wmma<2><<<dim3(32, 32), 256, 0, stream>>>(h2_16, w1_16, DMODEL,
        nullptr, nullptr, mlp_b1, nullptr, nullptr, nullptr, nullptr, nullptr, nullptr, m1_16);

    // MLP2 + gated residual -> d_out
    gemm_wmma<3><<<dim3(32, 8), 256, 0, stream>>>(m1_16, w2_16, DH4,
        mods, x2, mlp_b2, nullptr, nullptr, nullptr, nullptr, nullptr, out, nullptr);
}
